// TransformerBlock_8280696946968
// MI455X (gfx1250) — matrix operand order repaired and checked by reading, not compiled, not run on hardware
//
#include <hip/hip_runtime.h>
#include <hip/hip_bf16.h>

// ---------------------------------------------------------------------------
// MI455X (gfx1250) transformer block:
//   * bf16 end-to-end (fp32 residual stream), weights pre-transposed to [N][K]
//   * GEMM: TDM (tensor_load_to_lds) double-buffered staging, BK=64,
//           v_wmma_f32_16x16x32_bf16 compute, s_wait_tensorcnt sync
//   * Flash attention: TDM-staged Q/K tiles, online softmax, WMMA QK^T and PV
// ---------------------------------------------------------------------------

typedef __bf16 bf16_t;
typedef __attribute__((ext_vector_type(16))) __bf16 v16bf;
typedef __attribute__((ext_vector_type(8)))  __bf16 v8bf;
typedef __attribute__((ext_vector_type(8)))  float  v8f;
typedef __attribute__((ext_vector_type(4)))  unsigned int v4u;
typedef __attribute__((ext_vector_type(8)))  int v8i;
typedef __attribute__((ext_vector_type(4)))  int v4i;

#define DIM      2048
#define N_HEADS  16
#define HEAD_DIM 128
#define HIDDEN   8192
#define BATCH    2
#define SEQ      2048
#define EPS      1e-6f

#if defined(__gfx1250__) && __has_builtin(__builtin_amdgcn_tensor_load_to_lds) && \
    __has_builtin(__builtin_amdgcn_s_wait_tensorcnt)
#define HAVE_TDM 1
#else
#define HAVE_TDM 0
#endif

// Low 32 bits of a generic pointer to __shared__ = LDS byte address.
static __device__ __forceinline__ unsigned lds_off_of(const void* p) {
    return (unsigned)(unsigned long long)(uintptr_t)p;
}

// Issue one TDM 2D tile load: tile_rows x tile_cols bf16 elements from a
// row-major tensor with row_stride elements/row, packed row-major into LDS.
static __device__ __forceinline__ void tdm_load_2d(const bf16_t* gsrc,
                                                   unsigned lds_off,
                                                   int tile_rows, int tile_cols,
                                                   int row_stride,
                                                   unsigned tensor_rows) {
#if HAVE_TDM
    const unsigned long long ga = (unsigned long long)(uintptr_t)gsrc;
    const unsigned dim0 = (unsigned)row_stride;      // tensor_dim0 >= tile_cols
    v4u g0;
    g0.x = 1u;                                       // count=1 (valid user D#)
    g0.y = lds_off;                                  // lds_addr (bytes)
    g0.z = (unsigned)ga;                             // global_addr[31:0]
    g0.w = (unsigned)((ga >> 32) & 0x01FFFFFFull) | (2u << 30); // addr[56:32]|type=2
    v8i g1;
    g1[0] = (int)(1u << 16);                         // wg_mask=0, data_size=1 (2B)
    g1[1] = (int)((dim0 & 0xFFFFu) << 16);           // tensor_dim0[15:0]
    g1[2] = (int)((dim0 >> 16) | ((tensor_rows & 0xFFFFu) << 16)); // dim0 hi | dim1 lo
    g1[3] = (int)((tensor_rows >> 16) | ((unsigned)tile_cols << 16)); // dim1 hi | tile_dim0
    g1[4] = (int)((unsigned)tile_rows & 0xFFFFu);    // tile_dim1 | tile_dim2=0
    g1[5] = (int)dim0;                               // tensor_dim0_stride[31:0]
    g1[6] = 0;                                       // stride hi | dim1_stride lo
    g1[7] = 0;
    v4i z; z[0] = z[1] = z[2] = z[3] = 0;
    v8i z8; z8[0]=z8[1]=z8[2]=z8[3]=z8[4]=z8[5]=z8[6]=z8[7]=0;
    __builtin_amdgcn_tensor_load_to_lds(g0, g1, z, z, z8, 0);
#else
    (void)gsrc; (void)lds_off; (void)tile_rows; (void)tile_cols;
    (void)row_stride; (void)tensor_rows;
#endif
}

static __device__ __forceinline__ void tdm_wait() {
#if HAVE_TDM
    __builtin_amdgcn_s_wait_tensorcnt(0);
#endif
}

// Build a v16bf fragment from two contiguous 16-byte LDS chunks.
static __device__ __forceinline__ v16bf make_frag(const bf16_t* lo, const bf16_t* hi) {
    v8bf a = *(const v8bf*)lo;
    v8bf b = *(const v8bf*)hi;
    return __builtin_shufflevector(a, b, 0,1,2,3,4,5,6,7,8,9,10,11,12,13,14,15);
}

static __device__ __forceinline__ v8f wmma_bf16(v16bf a, v16bf b, v8f c) {
    return __builtin_amdgcn_wmma_f32_16x16x32_bf16(false, a, false, b, (short)0, c, false, false);
}

// ---------------------------------------------------------------------------
// Weight convert + transpose: W[K][N] fp32 -> WT[N][K] bf16 (LDS-tiled 32x32).
// ---------------------------------------------------------------------------
__global__ void __launch_bounds__(256)
transpose_convert_kernel(const float* __restrict__ W, bf16_t* __restrict__ WT,
                         int K, int N) {
    __shared__ float tile[32][33];
    const int tx = threadIdx.x & 31, ty = threadIdx.x >> 5;  // 32 x 8
    const int n0 = blockIdx.x * 32, k0 = blockIdx.y * 32;
    #pragma unroll
    for (int i = 0; i < 4; ++i) {
        int k = k0 + ty + i * 8;
        tile[ty + i * 8][tx] = W[(size_t)k * N + n0 + tx];
    }
    __syncthreads();
    #pragma unroll
    for (int i = 0; i < 4; ++i) {
        int n = n0 + ty + i * 8;
        WT[(size_t)n * K + k0 + tx] = (bf16_t)tile[tx][ty + i * 8];
    }
}

// ---------------------------------------------------------------------------
// RMSNorm: fp32 in, bf16 out. One block per row of length DIM.
// ---------------------------------------------------------------------------
__global__ void __launch_bounds__(256)
rmsnorm_kernel(const float* __restrict__ x, const float* __restrict__ w,
               bf16_t* __restrict__ out) {
    const int row = blockIdx.x;
    const float* xr = x + (size_t)row * DIM;
    bf16_t* orow = out + (size_t)row * DIM;

    float ss = 0.f;
    #pragma unroll
    for (int i = 0; i < DIM / 256; ++i) {
        float v = xr[threadIdx.x + i * 256];
        ss += v * v;
    }
    #pragma unroll
    for (int m = 16; m >= 1; m >>= 1) ss += __shfl_xor(ss, m, 32);

    __shared__ float partial[8];
    const int wave = threadIdx.x >> 5, lane = threadIdx.x & 31;
    if (lane == 0) partial[wave] = ss;
    __syncthreads();
    float tot = 0.f;
    #pragma unroll
    for (int i = 0; i < 8; ++i) tot += partial[i];

    const float scale = rsqrtf(tot / (float)DIM + EPS);
    #pragma unroll
    for (int i = 0; i < DIM / 256; ++i) {
        int c = threadIdx.x + i * 256;
        orow[c] = (bf16_t)(xr[c] * scale * w[c]);
    }
}

// ---------------------------------------------------------------------------
// Tiled GEMM: C[M,N] = A[M,K] @ BT[N,K]^T (+ fp32 residual). bf16 operands.
// Block 256 threads = 8 waves; tile 128x128; BK=64; double-buffered TDM.
// Wave w: rows 32*(w&3), cols 64*(w>>2) -> 2x4 WMMA tiles, 16 WMMA / k-step.
// ---------------------------------------------------------------------------
template <bool ADD_RES, typename CT>
__global__ void __launch_bounds__(256)
gemm_bf16(const bf16_t* __restrict__ A, const bf16_t* __restrict__ BT,
          const float* __restrict__ Res, CT* __restrict__ C,
          int M, int N, int K) {
    __shared__ __align__(16) bf16_t As[2][128][64];
    __shared__ __align__(16) bf16_t Bs[2][128][64];

    const int tid  = threadIdx.x;
    const int m0   = blockIdx.y * 128;
    const int n0   = blockIdx.x * 128;
    const int wave = tid >> 5, lane = tid & 31;
    const int h8   = lane >> 4, ln = lane & 15;
    const int wr   = wave & 3;
    const int wc   = wave >> 2;

    v8f acc[2][4];
    #pragma unroll
    for (int i = 0; i < 2; ++i)
        #pragma unroll
        for (int j = 0; j < 4; ++j)
            acc[i][j] = (v8f){0.f,0.f,0.f,0.f,0.f,0.f,0.f,0.f};

#if !HAVE_TDM
    auto stage = [&](int buf, int k0) {
        #pragma unroll
        for (int it = 0; it < 4; ++it) {
            int idx = tid + it * 256;        // 1024 16B chunks per tile
            int r = idx >> 3, c = (idx & 7) * 8;
            *(uint4*)&As[buf][r][c] = *(const uint4*)&A[(size_t)(m0 + r) * K + k0 + c];
            *(uint4*)&Bs[buf][r][c] = *(const uint4*)&BT[(size_t)(n0 + r) * K + k0 + c];
        }
    };
#endif

    // Prologue: stage k-block 0 into buffer 0.
#if HAVE_TDM
    if (tid == 0) {
        tdm_load_2d(A  + (size_t)m0 * K, lds_off_of(&As[0][0][0]), 128, 64, K, (unsigned)M);
        tdm_load_2d(BT + (size_t)n0 * K, lds_off_of(&Bs[0][0][0]), 128, 64, K, (unsigned)N);
    }
#else
    stage(0, 0);
#endif

    const int nk = K / 64;
    for (int kb = 0; kb < nk; ++kb) {
        const int buf = kb & 1;
#if HAVE_TDM
        if (tid == 0) tdm_wait();
#endif
        __syncthreads();   // buf ready; previous readers of buf^1 done
        if (kb + 1 < nk) {
#if HAVE_TDM
            if (tid == 0) {
                const int k1 = (kb + 1) * 64;
                tdm_load_2d(A  + (size_t)m0 * K + k1, lds_off_of(&As[buf ^ 1][0][0]),
                            128, 64, K, (unsigned)M);
                tdm_load_2d(BT + (size_t)n0 * K + k1, lds_off_of(&Bs[buf ^ 1][0][0]),
                            128, 64, K, (unsigned)N);
            }
#else
            stage(buf ^ 1, (kb + 1) * 64);
#endif
        }
        // Compute on buf: 2 kc chunks of 32, 2x4 tiles.
        #pragma unroll
        for (int kc = 0; kc < 2; ++kc) {
            v16bf af[2], bfv[4];
            #pragma unroll
            for (int i = 0; i < 2; ++i) {
                const bf16_t* rp = &As[buf][wr * 32 + i * 16 + ln][kc * 32];
                af[i] = make_frag(rp + h8 * 8, rp + 16 + h8 * 8);
            }
            #pragma unroll
            for (int j = 0; j < 4; ++j) {
                const bf16_t* cp = &Bs[buf][wc * 64 + j * 16 + ln][kc * 32];
                bfv[j] = make_frag(cp + h8 * 8, cp + 16 + h8 * 8);
            }
            #pragma unroll
            for (int i = 0; i < 2; ++i)
                #pragma unroll
                for (int j = 0; j < 4; ++j)
                    acc[i][j] = wmma_bf16(af[i], bfv[j], acc[i][j]);
        }
    }

    // Epilogue: C/D layout -> lane holds col (lane&15), rows h8*8 + i.
    #pragma unroll
    for (int i2 = 0; i2 < 2; ++i2) {
        #pragma unroll
        for (int j = 0; j < 4; ++j) {
            const int col   = n0 + wc * 64 + j * 16 + ln;
            const int rbase = m0 + wr * 32 + i2 * 16 + h8 * 8;
            #pragma unroll
            for (int i = 0; i < 8; ++i) {
                size_t off = (size_t)(rbase + i) * N + col;
                float vv = acc[i2][j][i];
                if (ADD_RES) vv += Res[off];
                C[off] = (CT)vv;
            }
        }
    }
}

// ---------------------------------------------------------------------------
// RoPE in place on bf16 (B, S, H, HEAD_DIM).
// ---------------------------------------------------------------------------
__global__ void rope_kernel(bf16_t* __restrict__ t, const float* __restrict__ fc,
                            const float* __restrict__ fs) {
    size_t id = (size_t)blockIdx.x * blockDim.x + threadIdx.x;
    if (id >= (size_t)BATCH * SEQ * N_HEADS * (HEAD_DIM / 2)) return;
    const int    i    = (int)(id & 63);
    const size_t rest = id >> 6;
    const int    h    = (int)(rest & (N_HEADS - 1));
    const size_t bs   = rest >> 4;
    const int    s    = (int)(bs & (SEQ - 1));
    const float c  = fc[s * 64 + i];
    const float sn = fs[s * 64 + i];
    const size_t base = (bs * N_HEADS + h) * HEAD_DIM + 2 * i;
    const float a = (float)t[base], b = (float)t[base + 1];
    t[base]     = (bf16_t)(a * c - b * sn);
    t[base + 1] = (bf16_t)(a * sn + b * c);
}

// ---------------------------------------------------------------------------
// Flash attention (causal), bf16 q/k/v/out. Grid: (SEQ/64, H, B), 128 thr.
// Q/K tiles staged via TDM; V staged manually (transposed layout [d][j]).
// ---------------------------------------------------------------------------
__global__ void __launch_bounds__(128)
flash_attn_kernel(const bf16_t* __restrict__ q, const bf16_t* __restrict__ k,
                  const bf16_t* __restrict__ v, bf16_t* __restrict__ out) {
    __shared__ __align__(16) bf16_t Qs[64][128];
    __shared__ __align__(16) bf16_t Ks[64][128];
    __shared__ __align__(16) bf16_t Vs[128][64];      // [d][j]
    __shared__ __align__(16) bf16_t Ps[4][16][64];    // per-wave P staging

    const int tid  = threadIdx.x;
    const int q0   = blockIdx.x * 64;
    const int head = blockIdx.y;
    const int b    = blockIdx.z;
    const int wave = tid >> 5, lane = tid & 31;
    const int h8   = lane >> 4, ln = lane & 15;
    const float scale = 0.08838834764831845f;          // 1/sqrt(128)
    const int row_stride = N_HEADS * HEAD_DIM;         // 2048 elements

    // Stage Q tile (64 x 128 bf16).
#if HAVE_TDM
    if (tid == 0) {
        tdm_load_2d(q + (((size_t)b * SEQ + q0) * N_HEADS + head) * HEAD_DIM,
                    lds_off_of(&Qs[0][0]), 64, 128, row_stride, (unsigned)SEQ);
        tdm_wait();
    }
#else
    #pragma unroll
    for (int it = 0; it < 8; ++it) {
        int idx = tid + it * 128;          // 1024 16B chunks
        int r = idx >> 4, c = (idx & 15) * 8;
        *(uint4*)&Qs[r][c] =
            *(const uint4*)&q[(((size_t)b * SEQ + q0 + r) * N_HEADS + head) * HEAD_DIM + c];
    }
#endif
    __syncthreads();

    v16bf aq[4];
    #pragma unroll
    for (int kc = 0; kc < 4; ++kc) {
        const bf16_t* rp = &Qs[wave * 16 + ln][kc * 32];
        aq[kc] = make_frag(rp + h8 * 8, rp + 16 + h8 * 8);
    }

    float m_i[8], l_i[8];
    v8f oacc[8];
    #pragma unroll
    for (int i = 0; i < 8; ++i) {
        m_i[i] = -3.0e38f; l_i[i] = 0.f;
        oacc[i] = (v8f){0.f,0.f,0.f,0.f,0.f,0.f,0.f,0.f};
    }

    for (int j0 = 0; j0 <= q0; j0 += 64) {
        __syncthreads();   // previous tile's consumers done
        // K tile via TDM (or cooperative copy); V tile manual transpose.
#if HAVE_TDM
        if (tid == 0) {
            tdm_load_2d(k + (((size_t)b * SEQ + j0) * N_HEADS + head) * HEAD_DIM,
                        lds_off_of(&Ks[0][0]), 64, 128, row_stride, (unsigned)SEQ);
        }
#else
        #pragma unroll
        for (int it = 0; it < 8; ++it) {
            int idx = tid + it * 128;
            int r = idx >> 4, c = (idx & 15) * 8;
            *(uint4*)&Ks[r][c] =
                *(const uint4*)&k[(((size_t)b * SEQ + j0 + r) * N_HEADS + head) * HEAD_DIM + c];
        }
#endif
        #pragma unroll
        for (int it = 0; it < 8; ++it) {
            int idx = tid + it * 128;       // 1024 8-elem chunks
            int r = idx >> 4, d0 = (idx & 15) * 8;
            v8bf g = *(const v8bf*)&v[(((size_t)b * SEQ + j0 + r) * N_HEADS + head) * HEAD_DIM + d0];
            #pragma unroll
            for (int e = 0; e < 8; ++e) Vs[d0 + e][r] = g[e];
        }
#if HAVE_TDM
        if (tid == 0) tdm_wait();
#endif
        __syncthreads();

        // S = Q @ K^T
        v8f sc[4];
        #pragma unroll
        for (int nt = 0; nt < 4; ++nt) {
            v8f s = (v8f){0.f,0.f,0.f,0.f,0.f,0.f,0.f,0.f};
            #pragma unroll
            for (int kc = 0; kc < 4; ++kc) {
                const bf16_t* cp = &Ks[nt * 16 + ln][kc * 32];
                v16bf bk = make_frag(cp + h8 * 8, cp + 16 + h8 * 8);
                s = wmma_bf16(aq[kc], bk, s);
            }
            sc[nt] = s;
        }

        const bool diag = (j0 == q0);
        #pragma unroll
        for (int nt = 0; nt < 4; ++nt) {
            const int j = j0 + nt * 16 + ln;
            #pragma unroll
            for (int i = 0; i < 8; ++i) {
                float val = sc[nt][i] * scale;
                if (diag) {
                    int qi = q0 + wave * 16 + h8 * 8 + i;
                    if (j > qi) val = -3.0e38f;
                }
                sc[nt][i] = val;
            }
        }

        // Online softmax (row stats across 16-lane groups).
        #pragma unroll
        for (int i = 0; i < 8; ++i) {
            float rm = fmaxf(fmaxf(sc[0][i], sc[1][i]), fmaxf(sc[2][i], sc[3][i]));
            rm = fmaxf(rm, __shfl_xor(rm, 1, 32));
            rm = fmaxf(rm, __shfl_xor(rm, 2, 32));
            rm = fmaxf(rm, __shfl_xor(rm, 4, 32));
            rm = fmaxf(rm, __shfl_xor(rm, 8, 32));
            const float mnew  = fmaxf(m_i[i], rm);
            const float alpha = __expf(m_i[i] - mnew);
            float rs = 0.f;
            #pragma unroll
            for (int nt = 0; nt < 4; ++nt) {
                float p = __expf(sc[nt][i] - mnew);
                sc[nt][i] = p;
                rs += p;
            }
            rs += __shfl_xor(rs, 1, 32);
            rs += __shfl_xor(rs, 2, 32);
            rs += __shfl_xor(rs, 4, 32);
            rs += __shfl_xor(rs, 8, 32);
            l_i[i] = l_i[i] * alpha + rs;
            m_i[i] = mnew;
            #pragma unroll
            for (int dt = 0; dt < 8; ++dt) oacc[dt][i] *= alpha;
            #pragma unroll
            for (int nt = 0; nt < 4; ++nt)
                Ps[wave][h8 * 8 + i][nt * 16 + ln] = (bf16_t)sc[nt][i];
        }

        // O += P @ V (per-wave Ps; same-wave LDS ops are in order).
        v16bf ap[2];
        #pragma unroll
        for (int kc = 0; kc < 2; ++kc) {
            const bf16_t* rp = &Ps[wave][ln][kc * 32];
            ap[kc] = make_frag(rp + h8 * 8, rp + 16 + h8 * 8);
        }
        #pragma unroll
        for (int dt = 0; dt < 8; ++dt) {
            #pragma unroll
            for (int kc = 0; kc < 2; ++kc) {
                const bf16_t* cp = &Vs[dt * 16 + ln][kc * 32];
                v16bf bv = make_frag(cp + h8 * 8, cp + 16 + h8 * 8);
                oacc[dt] = wmma_bf16(ap[kc], bv, oacc[dt]);
            }
        }
    }

    #pragma unroll
    for (int dt = 0; dt < 8; ++dt) {
        #pragma unroll
        for (int i = 0; i < 8; ++i) {
            const int qi = q0 + wave * 16 + h8 * 8 + i;
            const float inv = 1.0f / l_i[i];
            out[((size_t)b * SEQ + qi) * DIM + head * HEAD_DIM + dt * 16 + ln] =
                (bf16_t)(oacc[dt][i] * inv);
        }
    }
}

// ---------------------------------------------------------------------------
// ff1 = silu(ff1) * ff3 (bf16, in place on ff1)
// ---------------------------------------------------------------------------
__global__ void silu_mul_kernel(bf16_t* __restrict__ a, const bf16_t* __restrict__ b,
                                size_t n) {
    size_t id = (size_t)blockIdx.x * blockDim.x + threadIdx.x;
    if (id < n) {
        float x = (float)a[id];
        float s = x / (1.f + __expf(-x));
        a[id] = (bf16_t)(s * (float)b[id]);
    }
}

// ---------------------------------------------------------------------------
extern "C" void kernel_launch(void* const* d_in, const int* in_sizes, int n_in,
                              void* d_out, int out_size, void* d_ws, size_t ws_size,
                              hipStream_t stream) {
    (void)in_sizes; (void)n_in; (void)out_size; (void)ws_size;
    const float* x   = (const float*)d_in[0];
    const float* fc  = (const float*)d_in[1];
    const float* fs  = (const float*)d_in[2];
    /* d_in[3] = mask: causal mask applied analytically */
    const float* wq  = (const float*)d_in[4];
    const float* wk  = (const float*)d_in[5];
    const float* wv  = (const float*)d_in[6];
    const float* wo  = (const float*)d_in[7];
    const float* w1  = (const float*)d_in[8];
    const float* w2  = (const float*)d_in[9];
    const float* w3  = (const float*)d_in[10];
    const float* anw = (const float*)d_in[11];
    const float* fnw = (const float*)d_in[12];
    float* outp = (float*)d_out;

    const size_t ROWS = (size_t)BATCH * SEQ;            // 4096
    const size_t DD   = (size_t)DIM * DIM;              // 4.19M
    const size_t DH   = (size_t)DIM * HIDDEN;           // 16.8M
    const size_t RD   = ROWS * DIM;                     // 8.39M
    const size_t RH   = ROWS * HIDDEN;                  // 33.6M

    bf16_t* wb  = (bf16_t*)d_ws;
    bf16_t* wqT = wb;            bf16_t* wkT = wqT + DD;
    bf16_t* wvT = wkT + DD;      bf16_t* woT = wvT + DD;
    bf16_t* w1T = woT + DD;      bf16_t* w3T = w1T + DH;
    bf16_t* w2T = w3T + DH;
    bf16_t* xnb = w2T + DH;      // also reused as attn output
    bf16_t* qb  = xnb + RD;
    bf16_t* kb  = qb  + RD;
    bf16_t* vb  = kb  + RD;
    bf16_t* hnb = vb  + RD;
    bf16_t* f1b = hnb + RD;
    bf16_t* f3b = f1b + RH;
    float*  hb  = (float*)(f3b + RH);   // fp32 residual stream

    const dim3 tb(256);
    // Weight convert + transpose: W[K][N] -> WT[N][K] bf16.
    transpose_convert_kernel<<<dim3(DIM / 32, DIM / 32), tb, 0, stream>>>(wq, wqT, DIM, DIM);
    transpose_convert_kernel<<<dim3(DIM / 32, DIM / 32), tb, 0, stream>>>(wk, wkT, DIM, DIM);
    transpose_convert_kernel<<<dim3(DIM / 32, DIM / 32), tb, 0, stream>>>(wv, wvT, DIM, DIM);
    transpose_convert_kernel<<<dim3(DIM / 32, DIM / 32), tb, 0, stream>>>(wo, woT, DIM, DIM);
    transpose_convert_kernel<<<dim3(HIDDEN / 32, DIM / 32), tb, 0, stream>>>(w1, w1T, DIM, HIDDEN);
    transpose_convert_kernel<<<dim3(HIDDEN / 32, DIM / 32), tb, 0, stream>>>(w3, w3T, DIM, HIDDEN);
    transpose_convert_kernel<<<dim3(DIM / 32, HIDDEN / 32), tb, 0, stream>>>(w2, w2T, HIDDEN, DIM);

    const dim3 gsq(DIM / 128, (unsigned)(ROWS / 128));      // (16, 32)
    const dim3 gwide(HIDDEN / 128, (unsigned)(ROWS / 128)); // (64, 32)

    // 1) xn = rmsnorm(x) * attn_norm_w   (bf16)
    rmsnorm_kernel<<<(unsigned)ROWS, 256, 0, stream>>>(x, anw, xnb);
    // 2) q/k/v projections (bf16 out)
    gemm_bf16<false, bf16_t><<<gsq, tb, 0, stream>>>(xnb, wqT, nullptr, qb, (int)ROWS, DIM, DIM);
    gemm_bf16<false, bf16_t><<<gsq, tb, 0, stream>>>(xnb, wkT, nullptr, kb, (int)ROWS, DIM, DIM);
    gemm_bf16<false, bf16_t><<<gsq, tb, 0, stream>>>(xnb, wvT, nullptr, vb, (int)ROWS, DIM, DIM);
    // 3) RoPE on q, k
    {
        size_t n = ROWS * N_HEADS * (HEAD_DIM / 2);
        rope_kernel<<<(unsigned)((n + 255) / 256), 256, 0, stream>>>(qb, fc, fs);
        rope_kernel<<<(unsigned)((n + 255) / 256), 256, 0, stream>>>(kb, fc, fs);
    }
    // 4) flash attention -> reuse xnb as attn output
    {
        dim3 gfa(SEQ / 64, N_HEADS, BATCH);
        flash_attn_kernel<<<gfa, 128, 0, stream>>>(qb, kb, vb, xnb);
    }
    // 5) h = x + attn @ wo   (fp32 residual stream)
    gemm_bf16<true, float><<<gsq, tb, 0, stream>>>(xnb, woT, x, hb, (int)ROWS, DIM, DIM);
    // 6) hn = rmsnorm(h) * ffn_norm_w
    rmsnorm_kernel<<<(unsigned)ROWS, 256, 0, stream>>>(hb, fnw, hnb);
    // 7) ff1 = hn @ w1 ; ff3 = hn @ w3   (bf16)
    gemm_bf16<false, bf16_t><<<gwide, tb, 0, stream>>>(hnb, w1T, nullptr, f1b, (int)ROWS, HIDDEN, DIM);
    gemm_bf16<false, bf16_t><<<gwide, tb, 0, stream>>>(hnb, w3T, nullptr, f3b, (int)ROWS, HIDDEN, DIM);
    // 8) ff1 = silu(ff1) * ff3
    {
        size_t n = RH;
        silu_mul_kernel<<<(unsigned)((n + 255) / 256), 256, 0, stream>>>(f1b, f3b, n);
    }
    // 9) out = h + ff1 @ w2  (fp32)
    gemm_bf16<true, float><<<gsq, tb, 0, stream>>>(f1b, w2T, hb, outp, (int)ROWS, DIM, HIDDEN);
}
